// LearnableWatershedWithRNN_326417515004
// MI455X (gfx1250) — matrix operand order repaired and checked by reading, not compiled, not run on hardware
//
#include <hip/hip_runtime.h>
#include <hip/hip_bf16.h>

typedef __attribute__((ext_vector_type(16))) _Float16 v16h;
typedef __attribute__((ext_vector_type(8)))  _Float16 v8h;
typedef __attribute__((ext_vector_type(8)))  float    v8f;

#define Bsz   32
#define Hh    96
#define Ww    96
#define Lpix  (Hh*Ww)          // 9216
#define ITERS 50
#define KPAD  160              // 144 real K (16ch*9tap) padded to 5 chunks of 32

union F16x16 { v16h v; v8h h[2]; };

// ---------------------------------------------------------------- helpers
__device__ __forceinline__ float fast_sig(float x) {
    float e = __builtin_amdgcn_exp2f(-1.4426950408889634f * x);
    return __builtin_amdgcn_rcpf(1.0f + e);
}
__device__ __forceinline__ float fast_tanh(float x) {
    float e = __builtin_amdgcn_exp2f(2.8853900817779268f * x); // exp(2x)
    return 1.0f - 2.0f * __builtin_amdgcn_rcpf(e + 1.0f);
}

// ---------------------------------------------------------------- 1) grayscale
__global__ void k_gray(const float* __restrict__ img, float* __restrict__ gray) {
    int idx = blockIdx.x * blockDim.x + threadIdx.x;
    if (idx >= Bsz * Lpix) return;
    int b = idx / Lpix, pix = idx % Lpix;
    const float* p = img + (long)b * 3 * Lpix + pix;
    gray[idx] = 0.2989f * p[0] + 0.587f * p[Lpix] + 0.114f * p[2 * Lpix];
}

// ---------------------------------------------------------------- 2) Sobel -> G -> seq (G half of the interleaved raw-reshape view)
__global__ void k_sobel(const float* __restrict__ gray,
                        const float* __restrict__ Kx, const float* __restrict__ Ky,
                        float* __restrict__ seq) {
    int idx = blockIdx.x * blockDim.x + threadIdx.x;
    if (idx >= Bsz * Lpix) return;
    int b = idx / Lpix, pix = idx % Lpix;
    int h = pix / Ww, w = pix % Ww;
    float gx = 0.f, gy = 0.f;
    #pragma unroll
    for (int kh = 0; kh < 3; ++kh)
        #pragma unroll
        for (int kw = 0; kw < 3; ++kw) {
            int hh = h + kh - 1, ww = w + kw - 1;
            float v = (hh >= 0 && hh < Hh && ww >= 0 && ww < Ww)
                          ? gray[b * Lpix + hh * Ww + ww] : 0.f;
            gx += v * Kx[kh * 3 + kw];
            gy += v * Ky[kh * 3 + kw];
        }
    float G = sqrtf(gx * gx + gy * gy);
    int p = Lpix + pix, l = p >> 1, c = p & 1;     // flat p = L + pix
    seq[((long)l * Bsz + b) * 2 + c] = G;
}

// ---------------------------------------------------------------- 3) conv1 1->16 + relu, store f16 NHWC ([b][pix][ch], 32B/pixel)
__global__ void k_conv1(const float* __restrict__ gray,
                        const float* __restrict__ w1, const float* __restrict__ b1,
                        _Float16* __restrict__ act1) {
    int idx = blockIdx.x * blockDim.x + threadIdx.x;
    if (idx >= Bsz * Lpix) return;
    int b = idx / Lpix, pix = idx % Lpix;
    int h = pix / Ww, w = pix % Ww;
    float nb[9];
    #pragma unroll
    for (int t = 0; t < 9; ++t) {
        int hh = h + t / 3 - 1, ww = w + t % 3 - 1;
        nb[t] = (hh >= 0 && hh < Hh && ww >= 0 && ww < Ww)
                    ? gray[b * Lpix + hh * Ww + ww] : 0.f;
    }
    union { _Float16 hf[16]; uint4 q[2]; } o;
    #pragma unroll
    for (int oc = 0; oc < 16; ++oc) {
        float acc = b1[oc];
        #pragma unroll
        for (int t = 0; t < 9; ++t) acc += w1[oc * 9 + t] * nb[t];
        o.hf[oc] = (_Float16)fmaxf(acc, 0.f);
    }
    uint4* dst = (uint4*)(act1 + ((long)idx) * 16);
    dst[0] = o.q[0];
    dst[1] = o.q[1];
}

// ---------------------------------------------------------------- 4) pack w2 -> f16 [N=32][K=160], K = tap*16+ch, zero-padded
__global__ void k_wpack(const float* __restrict__ w2, _Float16* __restrict__ w2h) {
    int idx = blockIdx.x * blockDim.x + threadIdx.x;
    if (idx >= KPAD * 32) return;
    int o = idx / KPAD, k = idx % KPAD;
    float v = 0.f;
    if (k < 144) {
        int tap = k >> 4, ch = k & 15;
        v = w2[(o * 16 + ch) * 9 + tap];
    }
    w2h[o * KPAD + k] = (_Float16)v;
}

// ---------------------------------------------------------------- 5) conv2 16->32 implicit GEMM via WMMA, b128 LDS fragments
// one workgroup per (image b, row r); 4 waves; M tiles = 16-pixel spans, N tiles = 16 channels
__global__ void __launch_bounds__(128)
k_conv2_wmma(const _Float16* __restrict__ act1, const _Float16* __restrict__ w2h,
             const float* __restrict__ b2, float* __restrict__ act2) {
    __shared__ __align__(16) _Float16 a_s[3 * 98 * 16];   // [rr][cc][ch]   9408 B
    __shared__ __align__(16) _Float16 w_s[32 * KPAD];     // [n][k]        10240 B

    int b = blockIdx.x / Hh, r = blockIdx.x % Hh;
    int tid = threadIdx.x;

    // ---- weights: async copy global -> LDS (ASYNCcnt path), 640 x b128
    {
        uint64_t src = (uint64_t)(uintptr_t)w2h;
        uint32_t ldsbase = (uint32_t)(uintptr_t)(&w_s[0]);
        for (int j = tid; j < (32 * KPAD) / 8; j += 128) {
            uint32_t lds = ldsbase + j * 16;
            uint32_t off = (uint32_t)(j * 16);
            asm volatile("global_load_async_to_lds_b128 %0, %1, %2"
                         :: "v"(lds), "v"(off), "s"(src) : "memory");
        }
    }
    // ---- activation tile: channel-minor, contiguous uint4 copies (border -> 0)
    for (int j = tid; j < 3 * 98 * 2; j += 128) {
        int rr = j / (98 * 2), rem = j % (98 * 2);
        int cc = rem >> 1, cb = (rem & 1) * 8;
        int hh = r - 1 + rr, ww = cc - 1;
        uint4 v = make_uint4(0u, 0u, 0u, 0u);
        if (hh >= 0 && hh < Hh && ww >= 0 && ww < Ww)
            v = *(const uint4*)(act1 + ((long)b * Lpix + hh * Ww + ww) * 16 + cb);
        *(uint4*)(a_s + (rr * 98 + cc) * 16 + cb) = v;
    }
    asm volatile("s_wait_asynccnt 0" ::: "memory");
    __syncthreads();

    int wave = tid >> 5, lane = tid & 31;
    int hi = lane >> 4, lm = lane & 15;
    int n0 = (wave & 1) * 16;               // output-channel tile base

    // ---- B fragments: depend only on (chunk, lane); hoisted, reused across M tiles
    F16x16 bf[5];
    const _Float16* wb = &w_s[(n0 + lm) * KPAD + hi * 8];
    #pragma unroll
    for (int c = 0; c < 5; ++c) {
        bf[c].h[0] = *(const v8h*)(wb + c * 32);
        bf[c].h[1] = *(const v8h*)(wb + c * 32 + 16);
    }
    v8h zero8;
    #pragma unroll
    for (int e = 0; e < 8; ++e) zero8[e] = (_Float16)0;

    for (int t = 0; t < 3; ++t) {
        int c0 = ((wave >> 1) + 2 * t) * 16;     // 16-pixel span base
        v8f acc = {};
        #pragma unroll
        for (int c = 0; c < 5; ++c) {
            int te = 2 * c, to = 2 * c + 1;      // even/odd taps of this chunk
            F16x16 a;
            {
                int dr = te / 3, dc = te % 3;
                a.h[0] = *(const v8h*)(&a_s[((dr * 98) + (c0 + lm + dc)) * 16 + hi * 8]);
            }
            if (to < 9) {
                int dr = to / 3, dc = to % 3;
                a.h[1] = *(const v8h*)(&a_s[((dr * 98) + (c0 + lm + dc)) * 16 + hi * 8]);
            } else {
                a.h[1] = zero8;                  // K padding (tap 9)
            }
            acc = __builtin_amdgcn_wmma_f32_16x16x32_f16(
                false, a.v, false, bf[c].v, (short)0, acc, false, false);
        }
        // D: lane -> N, vgpr v -> M pixel; +bias +relu
        int oc = n0 + lm;
        float bias = b2[oc];
        #pragma unroll
        for (int v = 0; v < 8; ++v) {
            int cc = c0 + v + hi * 8;
            act2[((long)(b * 32 + oc)) * Lpix + r * Ww + cc] = fmaxf(acc[v] + bias, 0.f);
        }
    }
}

// ---------------------------------------------------------------- 6) conv3 32->3 + argmax -> markers half of seq
__global__ void k_conv3_argmax(const float* __restrict__ act2,
                               const float* __restrict__ w3, const float* __restrict__ b3,
                               float* __restrict__ seq) {
    int idx = blockIdx.x * blockDim.x + threadIdx.x;
    if (idx >= Bsz * Lpix) return;
    int b = idx / Lpix, pix = idx % Lpix;
    int h = pix / Ww, w = pix % Ww;
    float lg0 = b3[0], lg1 = b3[1], lg2 = b3[2];
    for (int ic = 0; ic < 32; ++ic) {
        float nb[9];
        #pragma unroll
        for (int t = 0; t < 9; ++t) {
            int hh = h + t / 3 - 1, ww = w + t % 3 - 1;
            nb[t] = (hh >= 0 && hh < Hh && ww >= 0 && ww < Ww)
                        ? act2[((long)(b * 32 + ic)) * Lpix + hh * Ww + ww] : 0.f;
        }
        const float* w0  = w3 + (0 * 32 + ic) * 9;
        const float* w1p = w3 + (1 * 32 + ic) * 9;
        const float* w2p = w3 + (2 * 32 + ic) * 9;
        #pragma unroll
        for (int t = 0; t < 9; ++t) {
            lg0 += w0[t] * nb[t];
            lg1 += w1p[t] * nb[t];
            lg2 += w2p[t] * nb[t];
        }
    }
    float m = 0.f, best = lg0;                 // softmax monotone; first-max ties
    if (lg1 > best) { best = lg1; m = 1.f; }
    if (lg2 > best) { m = 2.f; }
    int l = pix >> 1, c = pix & 1;             // markers plane: flat p = pix < L
    seq[((long)l * Bsz + b) * 2 + c] = m;
}

// ---------------------------------------------------------------- 7) GRU: 1 wave, lane = batch; 50*9216 dependent steps
__global__ void __launch_bounds__(32) k_gru(float* __restrict__ seq,
                                            const float* __restrict__ w_ih,
                                            const float* __restrict__ w_hh,
                                            const float* __restrict__ b_ih,
                                            const float* __restrict__ b_hh) {
    const int U = 8;
    int b = threadIdx.x;
    float wih[6][2], whh[6][2], bih[6], bhh[6];
    #pragma unroll
    for (int j = 0; j < 6; ++j) {
        wih[j][0] = w_ih[j * 2 + 0]; wih[j][1] = w_ih[j * 2 + 1];
        whh[j][0] = w_hh[j * 2 + 0]; whh[j][1] = w_hh[j * 2 + 1];
        bih[j] = b_ih[j]; bhh[j] = b_hh[j];
    }
    float2* sp = (float2*)seq;
    float h0 = 0.f, h1 = 0.f;

    float2 cur[U], nxt[U];
    #pragma unroll
    for (int u = 0; u < U; ++u) cur[u] = sp[(long)u * Bsz + b];

    int pos = 0;
    const long nblk = (long)ITERS * Lpix / U;
    for (long blk = 0; blk < nblk; ++blk) {
        int npos = pos + U; if (npos >= Lpix) npos = 0;
        int p2 = npos + U;  if (p2 >= Lpix) p2 = 0;
        __builtin_prefetch(&sp[(long)p2 * Bsz + b], 0, 1);     // global_prefetch_b8
        #pragma unroll
        for (int u = 0; u < U; ++u) nxt[u] = sp[(long)(npos + u) * Bsz + b];
        #pragma unroll
        for (int u = 0; u < U; ++u) {
            float x0 = cur[u].x, x1 = cur[u].y;
            float ir0 = fmaf(wih[0][0], x0, fmaf(wih[0][1], x1, bih[0]));
            float ir1 = fmaf(wih[1][0], x0, fmaf(wih[1][1], x1, bih[1]));
            float iz0 = fmaf(wih[2][0], x0, fmaf(wih[2][1], x1, bih[2]));
            float iz1 = fmaf(wih[3][0], x0, fmaf(wih[3][1], x1, bih[3]));
            float in0 = fmaf(wih[4][0], x0, fmaf(wih[4][1], x1, bih[4]));
            float in1 = fmaf(wih[5][0], x0, fmaf(wih[5][1], x1, bih[5]));
            float hr0 = fmaf(whh[0][0], h0, fmaf(whh[0][1], h1, bhh[0]));
            float hr1 = fmaf(whh[1][0], h0, fmaf(whh[1][1], h1, bhh[1]));
            float hz0 = fmaf(whh[2][0], h0, fmaf(whh[2][1], h1, bhh[2]));
            float hz1 = fmaf(whh[3][0], h0, fmaf(whh[3][1], h1, bhh[3]));
            float hn0 = fmaf(whh[4][0], h0, fmaf(whh[4][1], h1, bhh[4]));
            float hn1 = fmaf(whh[5][0], h0, fmaf(whh[5][1], h1, bhh[5]));
            float r0 = fast_sig(ir0 + hr0), r1 = fast_sig(ir1 + hr1);
            float z0 = fast_sig(iz0 + hz0), z1 = fast_sig(iz1 + hz1);
            float n0 = fast_tanh(fmaf(r0, hn0, in0));
            float n1 = fast_tanh(fmaf(r1, hn1, in1));
            h0 = fmaf(1.f - z0, n0, z0 * h0);
            h1 = fmaf(1.f - z1, n1, z1 * h1);
            float2 o; o.x = fmaxf(h0, 0.f); o.y = fmaxf(h1, 0.f);
            sp[(long)(pos + u) * Bsz + b] = o;
        }
        #pragma unroll
        for (int u = 0; u < U; ++u) cur[u] = nxt[u];
        pos = npos;
    }
}

// ---------------------------------------------------------------- 8) 1x1 output projection
__global__ void k_out(const float* __restrict__ seq,
                      const float* __restrict__ out_w, const float* __restrict__ out_b,
                      float* __restrict__ out) {
    int idx = blockIdx.x * blockDim.x + threadIdx.x;
    if (idx >= Bsz * Lpix) return;
    int b = idx / Lpix, pix = idx % Lpix;
    float f0 = seq[((long)pix * Bsz + b) * 2 + 0];
    float f1 = seq[((long)pix * Bsz + b) * 2 + 1];
    #pragma unroll
    for (int o = 0; o < 3; ++o)
        out[((long)(b * 3 + o)) * Lpix + pix] =
            fmaf(f0, out_w[o * 2 + 0], fmaf(f1, out_w[o * 2 + 1], out_b[o]));
}

// ---------------------------------------------------------------- launch
extern "C" void kernel_launch(void* const* d_in, const int* in_sizes, int n_in,
                              void* d_out, int out_size, void* d_ws, size_t ws_size,
                              hipStream_t stream) {
    (void)in_sizes; (void)n_in; (void)out_size; (void)ws_size;
    const float* img  = (const float*)d_in[0];
    const float* Kx   = (const float*)d_in[1];
    const float* Ky   = (const float*)d_in[2];
    const float* w1   = (const float*)d_in[3];
    const float* b1   = (const float*)d_in[4];
    const float* w2   = (const float*)d_in[5];
    const float* b2   = (const float*)d_in[6];
    const float* w3   = (const float*)d_in[7];
    const float* b3   = (const float*)d_in[8];
    const float* wih  = (const float*)d_in[9];
    const float* whh  = (const float*)d_in[10];
    const float* bih  = (const float*)d_in[11];
    const float* bhh  = (const float*)d_in[12];
    const float* outw = (const float*)d_in[13];
    const float* outb = (const float*)d_in[14];

    char* ws = (char*)d_ws;
    float*    gray  = (float*)(ws + 0);                 //  1,179,648 B
    _Float16* act1  = (_Float16*)(ws + 1179648);        //  9,437,184 B (NHWC)
    _Float16* w2h   = (_Float16*)(ws + 10616832);       //     10,240 B ([n][k])
    float*    act2  = (float*)(ws + 10627072);          // 37,748,736 B
    float*    seq   = (float*)(ws + 48375808);          //  2,359,296 B

    const int NPIX = Bsz * Lpix;
    dim3 blk(256), grd((NPIX + 255) / 256);

    k_gray<<<grd, blk, 0, stream>>>(img, gray);
    k_sobel<<<grd, blk, 0, stream>>>(gray, Kx, Ky, seq);
    k_conv1<<<grd, blk, 0, stream>>>(gray, w1, b1, act1);
    k_wpack<<<(KPAD * 32 + 255) / 256, 256, 0, stream>>>(w2, w2h);
    k_conv2_wmma<<<Bsz * Hh, 128, 0, stream>>>(act1, w2h, b2, act2);
    k_conv3_argmax<<<grd, blk, 0, stream>>>(act2, w3, b3, seq);
    k_gru<<<1, 32, 0, stream>>>(seq, wih, whh, bih, bhh);
    k_out<<<grd, blk, 0, stream>>>(seq, outw, outb, (float*)d_out);
}
